// UpdateTheta12E2E_67525475827841
// MI455X (gfx1250) — compile-verified, matrix operand order checked by reading, not run on hardware
//
#include <hip/hip_runtime.h>
#include <hip/hip_bf16.h>

// ---------------------------------------------------------------------------
// Problem constants (from the reference)
// ---------------------------------------------------------------------------
#define BB     256    // batch
#define DD     511    // feature dim
#define DP     512    // padded feature dim
#define HH     5120   // hidden dim
#define QCHUNK 4      // blocks per batch for the quad stream

typedef __attribute__((ext_vector_type(16))) _Float16 v16h;
typedef __attribute__((ext_vector_type(8)))  float    v8f;
typedef __attribute__((ext_vector_type(2)))  float    f32x2;

// ---------------------------------------------------------------------------
// fp32 -> fp16 convert with zero padding: dst is [Rp x Cp], src is [R x Cs]
// ---------------------------------------------------------------------------
__global__ __launch_bounds__(256) void cvt_pad_f16(
    const float* __restrict__ src, _Float16* __restrict__ dst,
    int R, int Cs, int Rp, int Cp)
{
    int idx = blockIdx.x * 256 + threadIdx.x;
    int total = Rp * Cp;
    if (idx >= total) return;
    int r = idx / Cp;
    int c = idx - r * Cp;
    float v = (r < R && c < Cs) ? src[(size_t)r * Cs + c] : 0.0f;
    dst[idx] = (_Float16)v;
}

// ---------------------------------------------------------------------------
// WMMA fragment loader (ISA 7.12.2, 16-bit A/B 16x32 layout):
//   lane l<16  : row = l,    v[0..7] = K{k0..k0+7},   v[8..15] = K{k0+16..k0+23}
//   lane l>=16 : row = l-16, v[0..7] = K{k0+8..k0+15}, v[8..15] = K{k0+24..k0+31}
// p must already point at (row, k0 + hsel*8); both 16B chunks are 16B aligned.
// ---------------------------------------------------------------------------
__device__ __forceinline__ v16h load_frag(const _Float16* __restrict__ p)
{
    union { uint4 q[2]; v16h v; } u;
    u.q[0] = *reinterpret_cast<const uint4*>(p);
    u.q[1] = *reinterpret_cast<const uint4*>(p + 16);
    return u.v;
}

// Epilogue: C/D layout (ISA 7.12.2): VGPR r -> M = r + 8*hsel, N = lane&15
template<int OUT_F16_RELU>
__device__ __forceinline__ void store_tile(
    v8f cc, int mt, int nt, int mm, int nn,
    const float* __restrict__ bias, void* __restrict__ C, int ldc, int n_real)
{
    int Ncol = nt + nn;
    float bv = (Ncol < n_real) ? bias[Ncol] : 0.0f;
#pragma unroll
    for (int rr = 0; rr < 8; ++rr) {
        float v = cc[rr] + bv;
        int Mrow = mt + mm + rr;
        if (OUT_F16_RELU) {
            v = fmaxf(v, 0.0f);
            ((_Float16*)C)[(size_t)Mrow * ldc + Ncol] = (_Float16)v;
        } else {
            ((float*)C)[(size_t)Mrow * ldc + Ncol] = v;
        }
    }
}

// ---------------------------------------------------------------------------
// WMMA GEMM:  C[M x N] = A[M x K] * B[N x K]^T (+bias, opt ReLU->f16)
//   A, B fp16 row-major (K contiguous).  8 waves/block: 2(M) x 4(N) waves,
//   each wave a 32x32 tile (2x2 v_wmma_f32_16x16x32_f16 fragments).
//   Block tile: 64(M) x 128(N).  grid.x = N/128, grid.y = M/64.
// ---------------------------------------------------------------------------
template<int OUT_F16_RELU>
__global__ __launch_bounds__(256) void gemm_f16_wmma(
    const _Float16* __restrict__ A,
    const _Float16* __restrict__ B,
    const float* __restrict__ bias,
    void* __restrict__ C,
    int K, int ldc, int n_real)
{
    const int lane = threadIdx.x & 31;
    const int wave = threadIdx.x >> 5;
    const int wm = wave >> 2;          // 0..1
    const int wn = wave & 3;           // 0..3
    const int m0 = blockIdx.y * 64 + wm * 32;
    const int n0 = blockIdx.x * 128 + wn * 32;
    const int r    = lane & 15;
    const int hsel = (lane >> 4) & 1;

    v8f c00 = {}, c01 = {}, c10 = {}, c11 = {};

    const _Float16* a0p = A + (size_t)(m0 + r) * K + hsel * 8;
    const _Float16* a1p = a0p + (size_t)16 * K;
    const _Float16* b0p = B + (size_t)(n0 + r) * K + hsel * 8;
    const _Float16* b1p = b0p + (size_t)16 * K;

    for (int k0 = 0; k0 < K; k0 += 32) {
        if (k0 + 64 < K) {  // pull the K-tile after next into WGP-level caches
            __builtin_prefetch(a0p + k0 + 64, 0, 3);
            __builtin_prefetch(b0p + k0 + 64, 0, 3);
        }
        v16h a0 = load_frag(a0p + k0);
        v16h a1 = load_frag(a1p + k0);
        v16h b0 = load_frag(b0p + k0);
        v16h b1 = load_frag(b1p + k0);
        c00 = __builtin_amdgcn_wmma_f32_16x16x32_f16(false, a0, false, b0,
                                                     (short)0, c00, false, false);
        c01 = __builtin_amdgcn_wmma_f32_16x16x32_f16(false, a0, false, b1,
                                                     (short)0, c01, false, false);
        c10 = __builtin_amdgcn_wmma_f32_16x16x32_f16(false, a1, false, b0,
                                                     (short)0, c10, false, false);
        c11 = __builtin_amdgcn_wmma_f32_16x16x32_f16(false, a1, false, b1,
                                                     (short)0, c11, false, false);
    }

    const int mm = hsel * 8;
    const int nn = lane & 15;
    store_tile<OUT_F16_RELU>(c00, m0,      n0,      mm, nn, bias, C, ldc, n_real);
    store_tile<OUT_F16_RELU>(c01, m0,      n0 + 16, mm, nn, bias, C, ldc, n_real);
    store_tile<OUT_F16_RELU>(c10, m0 + 16, n0,      mm, nn, bias, C, ldc, n_real);
    store_tile<OUT_F16_RELU>(c11, m0 + 16, n0 + 16, mm, nn, bias, C, ldc, n_real);
}

// ---------------------------------------------------------------------------
// quad partials: qpart[b*QCHUNK+c] = sum over rows [c*128, c*128+128) of
//   phi_b[i] * (M_b[i,:] . phi_b).
// One 256-thread block covers one 2KB row with a single coalesced float2
// non-temporal load per thread (267 MB stream -> TH=NT keeps it out of L2,
// preserving the reused GEMM working set).  Row stride 511 is odd, so row
// parity (b+i)&1 gives the 8-byte alignment head, handled scalar by t==0.
// ---------------------------------------------------------------------------
__global__ __launch_bounds__(256) void quad_kernel(
    const float* __restrict__ phi,      // [BB x DP] padded
    const float* __restrict__ invT,     // [BB x DD x DD]
    float* __restrict__ qpart)          // [BB * QCHUNK]
{
    __shared__ float sphi[DD];
    __shared__ float red[256];
    const int b = blockIdx.x / QCHUNK;
    const int c = blockIdx.x - b * QCHUNK;
    const int t = threadIdx.x;
    const float* p = phi + (size_t)b * DP;
    const float* M = invT + (size_t)b * DD * DD;

    for (int i = t; i < DD; i += 256) sphi[i] = p[i];
    __syncthreads();

    const int i0 = c * 128;
    const int i1 = (i0 + 128 < DD) ? (i0 + 128) : DD;

    float acc = 0.0f;
    for (int i = i0; i < i1; ++i) {
        const float pi = sphi[i];
        const float* __restrict__ row = M + (size_t)i * DD;
        const int head = (b + i) & 1;        // scalar head for 8B alignment
        float a = 0.0f;
        if (head && t == 0)
            a += __builtin_nontemporal_load(row) * sphi[0];
        const int j = head + 2 * t;
        if (j + 1 < DD) {
            f32x2 v = __builtin_nontemporal_load(
                reinterpret_cast<const f32x2*>(row + j));
            a += v.x * sphi[j] + v.y * sphi[j + 1];
        } else if (j < DD) {
            a += __builtin_nontemporal_load(row + j) * sphi[j];
        }
        acc += a * pi;
    }

    red[t] = acc;
    __syncthreads();
    for (int s = 128; s > 0; s >>= 1) {
        if (t < s) red[t] += red[t + s];
        __syncthreads();
    }
    if (t == 0) qpart[b * QCHUNK + c] = red[0];
}

// ---------------------------------------------------------------------------
// finish: quad = sum of partials; phi_n = phi * rsqrt(quad);
//         tiny 5-unit MLP; |.|; soft-threshold
// ---------------------------------------------------------------------------
__global__ __launch_bounds__(256) void finish_kernel(
    const float* __restrict__ phi,      // [BB x DP]
    const float* __restrict__ qpart,    // [BB * QCHUNK]
    const float* __restrict__ st_w1,    // [5]
    const float* __restrict__ st_b1,    // [5]
    const float* __restrict__ st_w2,    // [5]
    const float* __restrict__ st_b2,    // [1]
    float* __restrict__ out)            // [BB x DD]
{
    int idx = blockIdx.x * 256 + threadIdx.x;
    if (idx >= BB * DD) return;
    int b = idx / DD;
    int d = idx - b * DD;

    float q = qpart[b * QCHUNK + 0] + qpart[b * QCHUNK + 1]
            + qpart[b * QCHUNK + 2] + qpart[b * QCHUNK + 3];
    float pn = phi[(size_t)b * DP + d] * rsqrtf(q);

    float acc = st_b2[0];
#pragma unroll
    for (int k = 0; k < 5; ++k) {
        float g = fmaxf(pn * st_w1[k] + st_b1[k], 0.0f);
        acc += g * st_w2[k];
    }
    float lam = fabsf(acc) * 0.1f;
    float a = fabsf(pn) - lam;
    out[idx] = (a > 0.0f) ? copysignf(a, pn) : 0.0f;
}

// ---------------------------------------------------------------------------
// Host-side orchestration
// ---------------------------------------------------------------------------
extern "C" void kernel_launch(void* const* d_in, const int* in_sizes, int n_in,
                              void* d_out, int out_size, void* d_ws, size_t ws_size,
                              hipStream_t stream)
{
    // setup_inputs order:
    const float* theta_12     = (const float*)d_in[0];   // [256,511]
    // d_in[1] s_12, d_in[2] w_12 : unused by the reference
    const float* inv_Theta_11 = (const float*)d_in[3];   // [256,511,511]
    // d_in[4] Theta, d_in[5] col : unused by the reference
    const float* phi_w1       = (const float*)d_in[6];   // [5120,511]
    const float* phi_b1       = (const float*)d_in[7];   // [5120]
    const float* phi_w2       = (const float*)d_in[8];   // [511,5120]
    const float* phi_b2       = (const float*)d_in[9];   // [511]
    const float* st_w1        = (const float*)d_in[10];  // [5,1]
    const float* st_b1        = (const float*)d_in[11];  // [5]
    const float* st_w2        = (const float*)d_in[12];  // [1,5]
    const float* st_b2        = (const float*)d_in[13];  // [1]
    float* out = (float*)d_out;

    // Workspace layout (bytes)
    char* ws = (char*)d_ws;
    _Float16* wsA   = (_Float16*)(ws);                       // 256x512   f16
    _Float16* wsW1  = (_Float16*)(ws + 262144);              // 5120x512  f16
    _Float16* wsW2  = (_Float16*)(ws + 5505024);             // 512x5120  f16
    _Float16* wsH   = (_Float16*)(ws + 10747904);            // 256x5120  f16
    float*    wsPhi = (float*)   (ws + 13369344);            // 256x512   f32
    float*    wsQp  = (float*)   (ws + 13893632);            // 256x4     f32
    // total: 13897728 bytes

    // 1) converts (fp32 -> fp16, zero-padded)
    cvt_pad_f16<<<(BB * DP + 255) / 256, 256, 0, stream>>>(
        theta_12, wsA, BB, DD, BB, DP);
    cvt_pad_f16<<<(HH * DP + 255) / 256, 256, 0, stream>>>(
        phi_w1, wsW1, HH, DD, HH, DP);
    cvt_pad_f16<<<(DP * HH + 255) / 256, 256, 0, stream>>>(
        phi_w2, wsW2, DD, HH, DP, HH);

    // 2) GEMM1: h = relu(theta_12 @ phi_w1^T + b1)   [256 x 5120], f16 out
    {
        dim3 grid(HH / 128, BB / 64);
        gemm_f16_wmma<1><<<grid, 256, 0, stream>>>(
            wsA, wsW1, phi_b1, (void*)wsH, DP, HH, HH);
    }
    // 3) GEMM2: phi = h @ phi_w2^T + b2   [256 x 512(pad)], f32 out
    {
        dim3 grid(DP / 128, BB / 64);
        gemm_f16_wmma<0><<<grid, 256, 0, stream>>>(
            wsH, wsW2, phi_b2, (void*)wsPhi, HH, DP, DD);
    }
    // 4) quad partials: 4 blocks per batch over the 267 MB NT stream
    quad_kernel<<<BB * QCHUNK, 256, 0, stream>>>(wsPhi, inv_Theta_11, wsQp);

    // 5) reduce partials + normalize + st-MLP + soft threshold -> out [256x511]
    finish_kernel<<<(BB * DD + 255) / 256, 256, 0, stream>>>(
        wsPhi, wsQp, st_w1, st_b1, st_w2, st_b2, out);
}
